// GraphLearningLayer_38749194945137
// MI455X (gfx1250) — matrix operand order repaired and checked by reading, not compiled, not run on hardware
//
#include <hip/hip_runtime.h>
#include <math.h>

#define N_NODES 8192
#define IN_DIM  512
#define LDIM    256
#define NEG_SLOPE 0.01f
#define EPS_V     1e-10f

typedef _Float16 v16h __attribute__((ext_vector_type(16)));
typedef _Float16 v8h  __attribute__((ext_vector_type(8)));
typedef float    v8f  __attribute__((ext_vector_type(8)));

// ---------------- helpers ----------------

__device__ __forceinline__ v16h load_afrag(const _Float16* p) {
    // A fragment (16x32 f16): this lane needs halves [0..7] and [16..23] from p
    v8h lo = *(const v8h*)(p);
    v8h hi = *(const v8h*)(p + 16);
    return __builtin_shufflevector(lo, hi, 0,1,2,3,4,5,6,7,8,9,10,11,12,13,14,15);
}

__device__ __forceinline__ float halfmax16(float v) {
    v = fmaxf(v, __shfl_xor(v, 1, 16));
    v = fmaxf(v, __shfl_xor(v, 2, 16));
    v = fmaxf(v, __shfl_xor(v, 4, 16));
    v = fmaxf(v, __shfl_xor(v, 8, 16));
    return v;
}

__device__ __forceinline__ float halfsum16(float v) {
    v += __shfl_xor(v, 1, 16);
    v += __shfl_xor(v, 2, 16);
    v += __shfl_xor(v, 4, 16);
    v += __shfl_xor(v, 8, 16);
    return v;
}

__device__ __forceinline__ void atomicMaxF(float* addr, float val) {
    int* ai = (int*)addr;
    int old = __float_as_int(*addr);
    while (val > __int_as_float(old)) {
        int assumed = old;
        old = atomicCAS(ai, assumed, __float_as_int(val));
        if (old == assumed) break;
    }
}

// GEMM core: acc[r][c] = P[rowBase+16r .. +15][:] . P[colBase+16c .. +15][:]^T
__device__ __forceinline__ void gemm_tile(const _Float16* __restrict__ P,
                                          int rowBase, int colBase, int lane,
                                          v8f acc[2][2]) {
    const int m  = lane & 15;
    const int hs = lane >> 4;
    const _Float16* pa0 = P + (rowBase + m) * LDIM + hs * 8;
    const _Float16* pa1 = pa0 + 16 * LDIM;
    const _Float16* pb0 = P + (colBase + m) * LDIM + hs * 8;
    const _Float16* pb1 = pb0 + 16 * LDIM;
#pragma unroll
    for (int k0 = 0; k0 < LDIM; k0 += 32) {
        v16h a0 = load_afrag(pa0 + k0);
        v16h a1 = load_afrag(pa1 + k0);
        v16h b0 = load_afrag(pb0 + k0);
        v16h b1 = load_afrag(pb1 + k0);
        acc[0][0] = __builtin_amdgcn_wmma_f32_16x16x32_f16(false, a0, false, b0, (short)0, acc[0][0], false, false);
        acc[0][1] = __builtin_amdgcn_wmma_f32_16x16x32_f16(false, a0, false, b1, (short)0, acc[0][1], false, false);
        acc[1][0] = __builtin_amdgcn_wmma_f32_16x16x32_f16(false, a1, false, b0, (short)0, acc[1][0], false, false);
        acc[1][1] = __builtin_amdgcn_wmma_f32_16x16x32_f16(false, a1, false, b1, (short)0, acc[1][1], false, false);
    }
}

// ---------------- K0: sum(learn_w), init rowmax/rowsum ----------------

__global__ __launch_bounds__(256) void k_init(const float* __restrict__ learnw,
                                              float* __restrict__ sumw,
                                              float* __restrict__ rowmax,
                                              float* __restrict__ rowsum) {
    int gid = blockIdx.x * 256 + threadIdx.x;
    if (gid < N_NODES) {
        rowmax[gid] = -3.0e38f;
        rowsum[gid] = 0.0f;
    }
    if (blockIdx.x == 0) {
        __shared__ float red[256];
        red[threadIdx.x] = learnw[threadIdx.x];   // LDIM == 256 exactly
        __syncthreads();
        for (int s = 128; s > 0; s >>= 1) {
            if (threadIdx.x < s) red[threadIdx.x] += red[threadIdx.x + s];
            __syncthreads();
        }
        if (threadIdx.x == 0) *sumw = red[0];
    }
}

// ---------------- K1: P = normalize(x @ proj_w^T) * sqrt(learn_w), f16 ----------------

#define RPB 8   // rows per block

__global__ __launch_bounds__(256) void k_proj(const float* __restrict__ x,
                                              const float* __restrict__ projw,
                                              const float* __restrict__ learnw,
                                              _Float16* __restrict__ P) {
    __shared__ float xs[RPB][64];
    __shared__ float ssq[RPB];
    const int t = threadIdx.x;            // output dim 0..255
    const int rowBase = blockIdx.x * RPB;
    float h[RPB];
#pragma unroll
    for (int r = 0; r < RPB; ++r) h[r] = 0.0f;
    if (t < RPB) ssq[t] = 0.0f;

    for (int k0 = 0; k0 < IN_DIM; k0 += 64) {
        __syncthreads();
        for (int j = t; j < RPB * 64; j += 256) {
            int r = j >> 6, kk = j & 63;
            xs[r][kk] = x[(rowBase + r) * IN_DIM + k0 + kk];
        }
        __syncthreads();
        const float* w = projw + t * IN_DIM + k0;
        for (int kk = 0; kk < 64; ++kk) {
            float wv = w[kk];
#pragma unroll
            for (int r = 0; r < RPB; ++r) h[r] = fmaf(wv, xs[r][kk], h[r]);
        }
    }
#pragma unroll
    for (int r = 0; r < RPB; ++r) atomicAdd(&ssq[r], h[r] * h[r]);
    __syncthreads();
    const float sw = sqrtf(learnw[t]);
#pragma unroll
    for (int r = 0; r < RPB; ++r) {
        float inv = rsqrtf(ssq[r]);       // 1 / ||x_hat_row||
        P[(rowBase + r) * LDIM + t] = (_Float16)(h[r] * inv * sw);
    }
}

// ---------------- K2: scores -> leaky_relu -> row max ----------------

__global__ __launch_bounds__(256) void k_scores_rowmax(const _Float16* __restrict__ P,
                                                       const float* __restrict__ sumw_p,
                                                       float* __restrict__ rowmax) {
    const int lane = threadIdx.x & 31;
    const int wave = threadIdx.x >> 5;
    const int rowBase = blockIdx.y * 64  + (wave >> 2) * 32;
    const int colBase = blockIdx.x * 128 + (wave & 3) * 32;
    const float sumw = *sumw_p;

    v8f z = {0.f,0.f,0.f,0.f,0.f,0.f,0.f,0.f};
    v8f acc[2][2] = {{z, z}, {z, z}};
    gemm_tile(P, rowBase, colBase, lane, acc);

    const int hs = lane >> 4;
#pragma unroll
    for (int r = 0; r < 2; ++r) {
#pragma unroll
        for (int v = 0; v < 8; ++v) {
            float s0 = sumw - acc[r][0][v];
            float s1 = sumw - acc[r][1][v];
            s0 = (s0 > 0.f) ? s0 : s0 * NEG_SLOPE;
            s1 = (s1 > 0.f) ? s1 : s1 * NEG_SLOPE;
            float mx = halfmax16(fmaxf(s0, s1));
            if ((lane & 15) == 0)
                atomicMaxF(&rowmax[rowBase + r * 16 + hs * 8 + v], mx);
        }
    }
}

// ---------------- K3: e = adj * exp(s - rowmax) -> d_out, row sums ----------------

__global__ __launch_bounds__(256) void k_exp_out(const _Float16* __restrict__ P,
                                                 const float* __restrict__ sumw_p,
                                                 const float* __restrict__ adj,
                                                 const float* __restrict__ rowmax,
                                                 float* __restrict__ outp,
                                                 float* __restrict__ rowsum) {
    const int lane = threadIdx.x & 31;
    const int wave = threadIdx.x >> 5;
    const int rowBase = blockIdx.y * 64  + (wave >> 2) * 32;
    const int colBase = blockIdx.x * 128 + (wave & 3) * 32;
    const float sumw = *sumw_p;

    v8f z = {0.f,0.f,0.f,0.f,0.f,0.f,0.f,0.f};
    v8f acc[2][2] = {{z, z}, {z, z}};
    gemm_tile(P, rowBase, colBase, lane, acc);

    const int hs = lane >> 4;
    const int n  = lane & 15;
#pragma unroll
    for (int r = 0; r < 2; ++r) {
#pragma unroll
        for (int v = 0; v < 8; ++v) {
            const int row = rowBase + r * 16 + hs * 8 + v;
            const float mrow = rowmax[row];
            float psum = 0.0f;
#pragma unroll
            for (int c = 0; c < 2; ++c) {
                const int col = colBase + c * 16 + n;
                float s = sumw - acc[r][c][v];
                s = (s > 0.f) ? s : s * NEG_SLOPE;
                float e = adj[(size_t)row * N_NODES + col] * __expf(s - mrow);
                outp[(size_t)row * N_NODES + col] = e;
                psum += e;
            }
            psum = halfsum16(psum);
            if ((lane & 15) == 0) atomicAdd(&rowsum[row], psum);
        }
    }
}

// ---------------- K4: normalize rows + EPS ----------------

__global__ __launch_bounds__(256) void k_norm(float* __restrict__ outp,
                                              const float* __restrict__ rowsum) {
    size_t gid  = (size_t)blockIdx.x * 256 + threadIdx.x;
    size_t base = gid * 4;                       // 4 floats, same row (8192 % 4 == 0)
    int row = (int)(base >> 13);                 // / 8192
    float inv = 1.0f / rowsum[row];
    float4* p = (float4*)(outp + base);
    float4 vv = *p;
    vv.x = vv.x * inv + EPS_V;
    vv.y = vv.y * inv + EPS_V;
    vv.z = vv.z * inv + EPS_V;
    vv.w = vv.w * inv + EPS_V;
    *p = vv;
}

// ---------------- launch ----------------

extern "C" void kernel_launch(void* const* d_in, const int* in_sizes, int n_in,
                              void* d_out, int out_size, void* d_ws, size_t ws_size,
                              hipStream_t stream) {
    const float* x      = (const float*)d_in[0];   // [8192, 512]
    const float* adj    = (const float*)d_in[1];   // [8192, 8192]
    const float* projw  = (const float*)d_in[2];   // [256, 512]
    const float* learnw = (const float*)d_in[3];   // [256]
    float* outp = (float*)d_out;                   // [8192, 8192]

    char* ws = (char*)d_ws;
    _Float16* P   = (_Float16*)ws;                                   // 4 MB
    float* rowmax = (float*)(ws + (size_t)N_NODES * LDIM * 2);       // 32 KB
    float* rowsum = rowmax + N_NODES;                                // 32 KB
    float* sumw   = rowsum + N_NODES;                                // 4 B

    k_init<<<N_NODES / 256, 256, 0, stream>>>(learnw, sumw, rowmax, rowsum);
    k_proj<<<N_NODES / RPB, 256, 0, stream>>>(x, projw, learnw, P);

    dim3 g(N_NODES / 128, N_NODES / 64);
    k_scores_rowmax<<<g, 256, 0, stream>>>(P, sumw, rowmax);
    k_exp_out<<<g, 256, 0, stream>>>(P, sumw, adj, rowmax, outp, rowsum);

    k_norm<<<(size_t)N_NODES * (N_NODES / 4) / 256, 256, 0, stream>>>(outp, rowsum);
}
